// TransformerDecoderLayerGlobalImproved_9371618640158
// MI455X (gfx1250) — matrix operand order repaired and checked by reading, not compiled, not run on hardware
//
#include <hip/hip_runtime.h>

// ---------------------------------------------------------------------------
// Problem constants (from the reference)
// ---------------------------------------------------------------------------
#define BB   4
#define SS   2048
#define DD   1024
#define HH   16
#define FFD  4096
#define HDD  64

typedef __bf16 bf16;
typedef __attribute__((ext_vector_type(16))) __bf16 v16bf;
typedef __attribute__((ext_vector_type(8)))  __bf16 v8bf;
typedef __attribute__((ext_vector_type(8)))  float  v8f;

__device__ __forceinline__ v8f vzero8() {
  v8f z;
#pragma unroll
  for (int i = 0; i < 8; ++i) z[i] = 0.0f;
  return z;
}

// D = A(16x32 bf16) * B(32x16 bf16) + C, f32 accum
__device__ __forceinline__ v8f wmma_bf16(v16bf a, v16bf b, v8f c) {
  return __builtin_amdgcn_wmma_f32_16x16x32_bf16(
      /*neg_a=*/false, a, /*neg_b=*/false, b,
      /*c_mod=*/(short)0, c, /*reuse_a=*/false, /*reuse_b=*/false);
}

// ---------------------------------------------------------------------------
// Fragment loaders (wave32 WMMA VGPR layouts, ISA 7.12.2)
//
// A (16x32, 16-bit): lane L holds row M = L&15. Lanes 0-15 carry K=0..15,
// lanes 16-31 carry K=16..31 (same k order as B). Two contiguous 16-byte loads per lane.
// ---------------------------------------------------------------------------
__device__ __forceinline__ v16bf load_frag_a(const bf16* A, int ld, int row0,
                                             int k0, int lane) {
  const int r  = lane & 15;
  const int kb = (lane >> 4) << 4;       // 0 or 16
  const bf16* p = A + (size_t)(row0 + r) * ld + k0 + kb;
  v8bf lo = *(const v8bf*)p;             // K = kb .. kb+7
  v8bf hi = *(const v8bf*)(p + 8);       // K = kb+8 .. kb+15
  v16bf out;
#pragma unroll
  for (int i = 0; i < 8; ++i) { out[i] = lo[i]; out[i + 8] = hi[i]; }
  return out;
}

// B (32x16, 16-bit) taken as W^T where W is (N,K) row-major: lane L holds
// column N = L&15; lanes 0-15 carry K=0..15, lanes 16-31 carry K=16..31.
// One contiguous 32-byte load per lane (K-contiguous in W).
__device__ __forceinline__ v16bf load_frag_b(const bf16* W, int ld, int col0,
                                             int k0, int lane) {
  const int c  = lane & 15;
  const int kb = (lane >> 4) << 4;       // 0 or 16
  const bf16* p = W + (size_t)(col0 + c) * ld + k0 + kb;
  return *(const v16bf*)p;
}

// ---------------------------------------------------------------------------
// f32 -> bf16 conversion
// ---------------------------------------------------------------------------
__global__ void cvt_f32_bf16(const float* __restrict__ in, bf16* __restrict__ out,
                             int n) {
  int i = blockIdx.x * blockDim.x + threadIdx.x;
  if (i < n) out[i] = (bf16)in[i];
}

// ---------------------------------------------------------------------------
// Row LayerNorm: one 256-thread block per row of D=1024, bf16 output
// ---------------------------------------------------------------------------
__global__ __launch_bounds__(256) void ln_kernel(const float* __restrict__ x,
                                                 const float* __restrict__ g,
                                                 const float* __restrict__ bt,
                                                 bf16* __restrict__ y, int D) {
  const int row  = blockIdx.x;
  const int tid  = threadIdx.x;
  const int lane = tid & 31;
  const int wave = tid >> 5;
  const float* xr = x + (size_t)row * D;

  float s = 0.f, s2 = 0.f;
  for (int i = tid; i < D; i += 256) { float v = xr[i]; s += v; s2 += v * v; }
#pragma unroll
  for (int m = 16; m >= 1; m >>= 1) {
    s  += __shfl_xor(s,  m, 32);
    s2 += __shfl_xor(s2, m, 32);
  }
  __shared__ float ss[8], ss2[8];
  if (lane == 0) { ss[wave] = s; ss2[wave] = s2; }
  __syncthreads();
  if (wave == 0) {
    s  = (lane < 8) ? ss[lane]  : 0.f;
    s2 = (lane < 8) ? ss2[lane] : 0.f;
#pragma unroll
    for (int m = 4; m >= 1; m >>= 1) {
      s  += __shfl_xor(s,  m, 32);
      s2 += __shfl_xor(s2, m, 32);
    }
    if (lane == 0) { ss[0] = s; ss2[0] = s2; }
  }
  __syncthreads();
  const float mean = ss[0] / (float)D;
  const float var  = ss2[0] / (float)D - mean * mean;
  const float inv  = rsqrtf(var + 1e-5f);
  for (int i = tid; i < D; i += 256) {
    float v = (xr[i] - mean) * inv * g[i] + bt[i];
    y[(size_t)row * D + i] = (bf16)v;
  }
}

// ---------------------------------------------------------------------------
// bf16 WMMA GEMM: out = A[MxK] * W[NxK]^T (+bias) (relu) (+resid)
// Block = 256 threads = 8 waves (4x2 wave grid), 128x128 tile / block,
// 32x64 tile per wave (2x4 WMMA frags): each A fragment feeds 4 WMMAs ->
// 12 b128 loads per 8 WMMAs per K-step. M,N,K multiples of 128/128/32.
// vtmode==1: bf16 out stored as V-transposed [b,h,hd,s] for attention.
// ---------------------------------------------------------------------------
__global__ __launch_bounds__(256) void gemm_bf16_wmma(
    const bf16* __restrict__ A, const bf16* __restrict__ W,
    const float* __restrict__ bias, const float* resid,
    float* outF, bf16* outB,
    int M, int N, int K, int relu, int vtmode, int S_, int H_, int HD_) {
  const int lane = threadIdx.x & 31;
  const int wave = threadIdx.x >> 5;
  const int wm   = wave >> 1;                 // 0..3
  const int wn   = wave & 1;                  // 0..1
  const int m0   = blockIdx.y * 128 + wm * 32;
  const int n0   = blockIdx.x * 128 + wn * 64;

  v8f acc[2][4];
#pragma unroll
  for (int mi = 0; mi < 2; ++mi)
#pragma unroll
    for (int ni = 0; ni < 4; ++ni) acc[mi][ni] = vzero8();

  for (int k0 = 0; k0 < K; k0 += 32) {
    // Speculative prefetch of the next K-tiles (dropped silently if past end)
    __builtin_prefetch(A + (size_t)(m0 + (lane & 15)) * K + k0 + 32, 0, 1);
    __builtin_prefetch(W + (size_t)(n0 + (lane & 15)) * K + k0 + 32, 0, 1);

    v16bf a0 = load_frag_a(A, K, m0,      k0, lane);
    v16bf a1 = load_frag_a(A, K, m0 + 16, k0, lane);
    v16bf b0 = load_frag_b(W, K, n0,      k0, lane);
    v16bf b1 = load_frag_b(W, K, n0 + 16, k0, lane);
    v16bf b2 = load_frag_b(W, K, n0 + 32, k0, lane);
    v16bf b3 = load_frag_b(W, K, n0 + 48, k0, lane);
    acc[0][0] = wmma_bf16(a0, b0, acc[0][0]);
    acc[0][1] = wmma_bf16(a0, b1, acc[0][1]);
    acc[0][2] = wmma_bf16(a0, b2, acc[0][2]);
    acc[0][3] = wmma_bf16(a0, b3, acc[0][3]);
    acc[1][0] = wmma_bf16(a1, b0, acc[1][0]);
    acc[1][1] = wmma_bf16(a1, b1, acc[1][1]);
    acc[1][2] = wmma_bf16(a1, b2, acc[1][2]);
    acc[1][3] = wmma_bf16(a1, b3, acc[1][3]);
  }

  const int half = lane >> 4;
  const int cn   = lane & 15;
#pragma unroll
  for (int mi = 0; mi < 2; ++mi)
#pragma unroll
    for (int ni = 0; ni < 4; ++ni)
#pragma unroll
      for (int r = 0; r < 8; ++r) {
        const int m = m0 + mi * 16 + r + half * 8;  // C layout: M=r(+8), N=lane
        const int n = n0 + ni * 16 + cn;
        float v = acc[mi][ni][r];
        if (bias)  v += bias[n];
        if (relu)  v = fmaxf(v, 0.0f);
        if (resid) v += resid[(size_t)m * N + n];
        if (outF)  outF[(size_t)m * N + n] = v;
        if (outB) {
          if (vtmode) {
            const int bb = m / S_, s = m % S_;
            const int hh = n / HD_, hd = n % HD_;
            outB[(((size_t)bb * H_ + hh) * HD_ + hd) * S_ + s] = (bf16)v;
          } else {
            outB[(size_t)m * N + n] = (bf16)v;
          }
        }
      }
}

// ---------------------------------------------------------------------------
// Causal attention, flash-style online softmax. One wave per (b,h,16-query
// tile). Scores via WMMA (Q 16x64 x K^T 64x16), P staged through LDS and
// re-loaded in A layout, P·V via WMMA against V stored transposed [b,h,hd,s].
// ---------------------------------------------------------------------------
__global__ __launch_bounds__(32) void attn_kernel(const bf16* __restrict__ q,
                                                  const bf16* __restrict__ k,
                                                  const bf16* __restrict__ vt,
                                                  bf16* __restrict__ o) {
  const int lane = threadIdx.x;
  const int s0   = blockIdx.x * 16;
  const int h    = blockIdx.y;
  const int b    = blockIdx.z;
  const int half = lane >> 4;
  const int cn   = lane & 15;

  const bf16* qbase = q  + (size_t)b * SS * DD + h * HDD;           // [s, hd], ld=DD
  const bf16* kbase = k  + (size_t)b * SS * DD + h * HDD;           // [s, hd], ld=DD
  const bf16* vbase = vt + ((size_t)b * HH + h) * (size_t)HDD * SS; // [hd, s], ld=SS

  __shared__ bf16 psh[16 * 32];

  const v16bf qa0 = load_frag_a(qbase, DD, s0, 0,  lane);
  const v16bf qa1 = load_frag_a(qbase, DD, s0, 32, lane);

  v8f oacc[4];
#pragma unroll
  for (int ni = 0; ni < 4; ++ni) oacc[ni] = vzero8();
  float mx[8], ls[8];
#pragma unroll
  for (int r = 0; r < 8; ++r) { mx[r] = -3.0e38f; ls[r] = 0.0f; }

  for (int t0 = 0; t0 < s0 + 16; t0 += 32) {   // keys t0..t0+31 (over-mask ok)
    v16bf kf0a = load_frag_b(kbase, DD, t0,      0,  lane);
    v16bf kf0b = load_frag_b(kbase, DD, t0,      32, lane);
    v16bf kf1a = load_frag_b(kbase, DD, t0 + 16, 0,  lane);
    v16bf kf1b = load_frag_b(kbase, DD, t0 + 16, 32, lane);
    v8f sc0 = vzero8(), sc1 = vzero8();
    sc0 = wmma_bf16(qa0, kf0a, sc0);
    sc0 = wmma_bf16(qa1, kf0b, sc0);
    sc1 = wmma_bf16(qa0, kf1a, sc1);
    sc1 = wmma_bf16(qa1, kf1b, sc1);

    float alph[8];
#pragma unroll
    for (int r = 0; r < 8; ++r) {
      const int m  = r + half * 8;
      const int qi = s0 + m;
      float v0 = sc0[r] * 0.125f;              // 1/sqrt(HD)
      float v1 = sc1[r] * 0.125f;
      if (t0 + cn > qi)      v0 = -3.0e38f;    // causal mask
      if (t0 + 16 + cn > qi) v1 = -3.0e38f;
      float tm = fmaxf(v0, v1);
#pragma unroll
      for (int d = 1; d < 16; d <<= 1) tm = fmaxf(tm, __shfl_xor(tm, d, 32));
      const float mnew = fmaxf(mx[r], tm);
      const float a  = __expf(mx[r] - mnew);
      const float p0 = __expf(v0 - mnew);
      const float p1 = __expf(v1 - mnew);
      float ps = p0 + p1;
#pragma unroll
      for (int d = 1; d < 16; d <<= 1) ps += __shfl_xor(ps, d, 32);
      ls[r]  = ls[r] * a + ps;
      mx[r]  = mnew;
      alph[r] = a;
      psh[m * 32 + cn]      = (bf16)p0;        // P tile 16x32, row-major in LDS
      psh[m * 32 + 16 + cn] = (bf16)p1;
    }
#pragma unroll
    for (int ni = 0; ni < 4; ++ni)
#pragma unroll
      for (int r = 0; r < 8; ++r) oacc[ni][r] *= alph[r];

    __syncthreads();                           // LDS stores -> loads
    const v16bf pa = load_frag_a(psh, 32, 0, 0, lane);
#pragma unroll
    for (int ni = 0; ni < 4; ++ni) {
      v16bf vf = load_frag_b(vbase, SS, ni * 16, t0, lane);  // cols=hd, K=keys
      oacc[ni] = wmma_bf16(pa, vf, oacc[ni]);
    }
    __syncthreads();                           // loads done before overwrite
  }

#pragma unroll
  for (int ni = 0; ni < 4; ++ni)
#pragma unroll
    for (int r = 0; r < 8; ++r) {
      const int m = r + half * 8;
      const float v = oacc[ni][r] / ls[r];
      o[((size_t)b * SS + s0 + m) * DD + h * HDD + ni * 16 + cn] = (bf16)v;
    }
}

// ---------------------------------------------------------------------------
// Host orchestration
// ---------------------------------------------------------------------------
extern "C" void kernel_launch(void* const* d_in, const int* in_sizes, int n_in,
                              void* d_out, int out_size, void* d_ws, size_t ws_size,
                              hipStream_t stream) {
  (void)in_sizes; (void)n_in; (void)out_size; (void)ws_size;
  const float* tgt    = (const float*)d_in[0];
  const float* memory = (const float*)d_in[1];
  // d_in[2] is the tril mask; causality is applied analytically in attn_kernel.
  const float* Wq = (const float*)d_in[3];  const float* bq  = (const float*)d_in[4];
  const float* Wk = (const float*)d_in[5];  const float* bk  = (const float*)d_in[6];
  const float* Wv = (const float*)d_in[7];  const float* bv  = (const float*)d_in[8];
  const float* Wo = (const float*)d_in[9];  const float* bo  = (const float*)d_in[10];
  const float* Wg = (const float*)d_in[11]; const float* bg  = (const float*)d_in[12];
  const float* W1 = (const float*)d_in[13]; const float* b1  = (const float*)d_in[14];
  const float* W2 = (const float*)d_in[15]; const float* b2  = (const float*)d_in[16];
  const float* g1 = (const float*)d_in[17]; const float* be1 = (const float*)d_in[18];
  const float* g2 = (const float*)d_in[19]; const float* be2 = (const float*)d_in[20];
  float* out = (float*)d_out;

  const size_t NT = (size_t)BB * SS;   // 8192 token rows

  // Workspace partition, 256-byte aligned (~234 MB total)
  size_t off = 0;
  auto wsalloc = [&](size_t bytes) -> void* {
    void* p = (char*)d_ws + off;
    off += (bytes + 255) & ~(size_t)255;
    return p;
  };
  bf16* wq_b  = (bf16*)wsalloc((size_t)DD * DD * sizeof(bf16));
  bf16* wk_b  = (bf16*)wsalloc((size_t)DD * DD * sizeof(bf16));
  bf16* wv_b  = (bf16*)wsalloc((size_t)DD * DD * sizeof(bf16));
  bf16* wo_b  = (bf16*)wsalloc((size_t)DD * DD * sizeof(bf16));
  bf16* wg_b  = (bf16*)wsalloc((size_t)DD * DD * sizeof(bf16));
  bf16* w1_b  = (bf16*)wsalloc((size_t)FFD * DD * sizeof(bf16));
  bf16* w2_b  = (bf16*)wsalloc((size_t)DD * FFD * sizeof(bf16));
  bf16* mem_b = (bf16*)wsalloc(NT * DD * sizeof(bf16));
  bf16* x1_b  = (bf16*)wsalloc(NT * DD * sizeof(bf16));
  bf16* q_b   = (bf16*)wsalloc(NT * DD * sizeof(bf16));
  bf16* k_b   = (bf16*)wsalloc(NT * DD * sizeof(bf16));
  bf16* vt_b  = (bf16*)wsalloc(NT * DD * sizeof(bf16));   // [b,h,hd,s]
  bf16* at_b  = (bf16*)wsalloc(NT * DD * sizeof(bf16));
  float* x2   = (float*)wsalloc(NT * DD * sizeof(float));
  bf16* x2n_b = (bf16*)wsalloc(NT * DD * sizeof(bf16));
  bf16* h_b   = (bf16*)wsalloc(NT * FFD * sizeof(bf16));

  auto cvt = [&](const float* src, bf16* dst, size_t n) {
    cvt_f32_bf16<<<dim3((unsigned)((n + 255) / 256)), dim3(256), 0, stream>>>(
        src, dst, (int)n);
  };
  auto gemm = [&](const bf16* A, const bf16* Wm, const float* bias,
                  const float* resid, float* oF, bf16* oB,
                  int M, int N, int K, int relu, int vtmode) {
    gemm_bf16_wmma<<<dim3(N / 128, M / 128), dim3(256), 0, stream>>>(
        A, Wm, bias, resid, oF, oB, M, N, K, relu, vtmode, SS, HH, HDD);
  };

  // One-time precision down-conversion of weights + memory
  cvt(Wq, wq_b, (size_t)DD * DD);
  cvt(Wk, wk_b, (size_t)DD * DD);
  cvt(Wv, wv_b, (size_t)DD * DD);
  cvt(Wo, wo_b, (size_t)DD * DD);
  cvt(Wg, wg_b, (size_t)DD * DD);
  cvt(W1, w1_b, (size_t)FFD * DD);
  cvt(W2, w2_b, (size_t)DD * FFD);
  cvt(memory, mem_b, NT * DD);

  // tgt1 = LN(tgt)
  ln_kernel<<<dim3((unsigned)NT), dim3(256), 0, stream>>>(tgt, g1, be1, x1_b, DD);

  // x2 = tgt + memory @ Wg^T + bg
  gemm(mem_b, wg_b, bg, tgt, x2, nullptr, (int)NT, DD, DD, 0, 0);

  // Q, K, V projections (V written transposed for the P·V WMMA B operand)
  gemm(x1_b, wq_b, bq, nullptr, nullptr, q_b,  (int)NT, DD, DD, 0, 0);
  gemm(x1_b, wk_b, bk, nullptr, nullptr, k_b,  (int)NT, DD, DD, 0, 0);
  gemm(x1_b, wv_b, bv, nullptr, nullptr, vt_b, (int)NT, DD, DD, 0, 1);

  // Causal attention -> at_b (bf16, [b,s,d])
  attn_kernel<<<dim3(SS / 16, HH, BB), dim3(32), 0, stream>>>(q_b, k_b, vt_b, at_b);

  // x2 += attn @ Wo^T + bo   (in-place residual; each element R/W by one thread)
  gemm(at_b, wo_b, bo, x2, x2, nullptr, (int)NT, DD, DD, 0, 0);

  // tgt1 = LN(x2)
  ln_kernel<<<dim3((unsigned)NT), dim3(256), 0, stream>>>(x2, g2, be2, x2n_b, DD);

  // h = relu(tgt1 @ W1^T + b1)
  gemm(x2n_b, w1_b, b1, nullptr, nullptr, h_b, (int)NT, FFD, DD, 1, 0);

  // out = x2 + h @ W2^T + b2
  gemm(h_b, w2_b, b2, x2, out, nullptr, (int)NT, DD, FFD, 0, 0);
}
